// Attention_4140348473538
// MI455X (gfx1250) — matrix operand order repaired and checked by reading, not compiled, not run on hardware
//
#include <hip/hip_runtime.h>

// ---------------- CDNA5 wave32 WMMA types ----------------
typedef __attribute__((ext_vector_type(16))) __bf16 v16bf;
typedef __attribute__((ext_vector_type(8)))  float  v8f;
typedef __attribute__((ext_vector_type(4)))  unsigned int v4u;
typedef __attribute__((ext_vector_type(8)))  int  v8i;
typedef __attribute__((ext_vector_type(4)))  int  v4i;

#define WMMA_BF16(A, B, C) \
  __builtin_amdgcn_wmma_f32_16x16x32_bf16(false, (A), false, (B), (short)0, (C), false, false)

// Tensor Data Mover: arity differs between toolchains; the TDM header ships
// only with the 6-arg (amdgpu-toolchain / clang-23) flavor.
#if __has_include(<hip/amd_detail/amd_gfx1250_TDM.h>)
#define TDM_LOAD_TO_LDS(g0, g1, z4, z8) \
  __builtin_amdgcn_tensor_load_to_lds((g0), (g1), (z4), (z4), (z8), 0)
#else
#define TDM_LOAD_TO_LDS(g0, g1, z4, z8) \
  __builtin_amdgcn_tensor_load_to_lds((g0), (g1), (z4), (z4), 0)
#endif

__device__ __forceinline__ __bf16 f2bf(float f) {
  unsigned u = __float_as_uint(f);
  u = (u + 0x7FFFu + ((u >> 16) & 1u)) >> 16;   // round-to-nearest-even
  unsigned short s = (unsigned short)u;
  __bf16 r;
  __builtin_memcpy(&r, &s, 2);
  return r;
}

// A-fragment (16x32 bf16) element map:  lane half = lane>>4, element e in [0,16)
//   K = e + (e>=8 ? 8 : 0) + half*8     (ISA 7.12.2, 16-bit A 16x32)
__device__ __forceinline__ void a_he(int k, int& half, int& e) {
  if (k < 8)       { half = 0; e = k; }
  else if (k < 16) { half = 1; e = k - 8; }
  else if (k < 24) { half = 0; e = k - 8; }
  else             { half = 1; e = k - 16; }
}

// ---------------- problem constants ----------------
#define BB      2
#define SS      2048
#define DD      1024
#define NH      16
#define NKV     4
#define HD      64
#define WIN     512
#define BS      (BB * SS)      // 4096 token rows

// =====================================================================
// pack fp32 row-major [M,K] -> bf16 A-fragment tiles [M/16][K/32][32][16]
// =====================================================================
__global__ void pack_a_kernel(const float* __restrict__ src, __bf16* __restrict__ dst,
                              int M, int K) {
  int Kt = K >> 5;
  size_t t = (size_t)blockIdx.x * blockDim.x + threadIdx.x;
  if (t >= (size_t)M * K) return;
  int e    = (int)(t & 15);
  int lane = (int)((t >> 4) & 31);
  size_t r = t >> 9;
  int kt = (int)(r % Kt);
  int mt = (int)(r / Kt);
  int half = lane >> 4;
  int k = kt * 32 + e + (e >= 8 ? 8 : 0) + half * 8;
  int m = mt * 16 + (lane & 15);
  dst[t] = f2bf(src[(size_t)m * K + k]);
}

// =====================================================================
// pack fp32 row-major [K,N] -> bf16 B-fragment tiles [K/32][N/16][32][16]
// =====================================================================
__global__ void pack_b_kernel(const float* __restrict__ src, __bf16* __restrict__ dst,
                              int K, int N) {
  int Nt = N >> 4;
  size_t t = (size_t)blockIdx.x * blockDim.x + threadIdx.x;
  if (t >= (size_t)K * N) return;
  int e    = (int)(t & 15);
  int lane = (int)((t >> 4) & 31);
  size_t r = t >> 9;
  int nt = (int)(r % Nt);
  int kt = (int)(r / Nt);
  int k = kt * 32 + e + (e >= 8 ? 8 : 0) + (lane >> 4) * 8;
  int n = nt * 16 + (lane & 15);
  dst[t] = f2bf(src[(size_t)k * N + n]);
}

// =====================================================================
// pack V fp32 [BS, NKV*HD] -> bf16 B-frags Vp[b][kvh][S/32][4 dimtiles][32][16]
// =====================================================================
__global__ void pack_v_kernel(const float* __restrict__ v, __bf16* __restrict__ dst) {
  size_t total = (size_t)BB * NKV * SS * HD;
  size_t t = (size_t)blockIdx.x * blockDim.x + threadIdx.x;
  if (t >= total) return;
  int e    = (int)(t & 15);
  int lane = (int)((t >> 4) & 31);
  size_t r = t >> 9;
  int dt  = (int)(r & 3);      r >>= 2;
  int kt  = (int)(r % (SS / 32)); r /= (SS / 32);
  int kvh = (int)(r % NKV);
  int b   = (int)(r / NKV);
  int s = kt * 32 + e + (e >= 8 ? 8 : 0) + (lane >> 4) * 8;
  int d = dt * 16 + (lane & 15);
  dst[t] = f2bf(v[((size_t)(b * SS + s)) * (NKV * HD) + kvh * HD + d]);
}

// =====================================================================
// fused RMSNorm + RoPE, one wave per (b, s, head).
// =====================================================================
__global__ void norm_rope_pack_kernel(const float* __restrict__ src,
                                      const float* __restrict__ w,
                                      __bf16* __restrict__ dst,
                                      int heads, int is_q, float outscale) {
  int lane = threadIdx.x & 31;
  int wave = (int)((blockIdx.x * (size_t)blockDim.x + threadIdx.x) >> 5);
  int h = wave % heads;
  int s = (wave / heads) % SS;
  int b = wave / (heads * SS);
  int Dsrc = heads * HD;

  const float* row = src + (size_t)(b * SS + s) * Dsrc + h * HD;
  float x0 = row[lane];
  float x1 = row[lane + 32];

  float ss = x0 * x0 + x1 * x1;
#pragma unroll
  for (int off = 1; off < 32; off <<= 1) ss += __shfl_xor(ss, off, 32);
  float scale = rsqrtf(ss * (1.0f / (float)HD) + 1e-5f);

  float y0 = x0 * scale * w[lane];
  float y1 = x1 * scale * w[lane + 32];

  float inv = __powf(10000.0f, -((float)lane) * (1.0f / 32.0f));
  float ang = (float)s * inv;
  float c  = __cosf(ang);
  float sn = __sinf(ang);
  float n0 = __shfl_xor(y0, 1, 32);
  float n1 = __shfl_xor(y1, 1, 32);
  float r0 = (lane & 1) ? n0 : -n0;
  float r1 = (lane & 1) ? n1 : -n1;
  float z0 = (y0 * c + r0 * sn) * outscale;
  float z1 = (y1 * c + r1 * sn) * outscale;

  size_t base = ((size_t)(b * heads + h) * (SS / 16) + (s >> 4)) * 2 * 512;
  if (is_q) {
    int m = s & 15;
    int ha, e;
    a_he(lane, ha, e);
    dst[base +       (size_t)(ha * 16 + m) * 16 + e] = f2bf(z0);
    dst[base + 512 + (size_t)(ha * 16 + m) * 16 + e] = f2bf(z1);
  } else {
    int n = s & 15;
    int hb, e;
    a_he(lane, hb, e);
    dst[base +       (size_t)(hb * 16 + n) * 16 + e] = f2bf(z0);
    dst[base + 512 + (size_t)(hb * 16 + n) * 16 + e] = f2bf(z1);
  }
}

// =====================================================================
// bf16 WMMA GEMM with TDM-staged B panel.
// Block = 8 waves (2 x 4) covering a 64 x 128 output tile.
// Per 32-deep k-step the 8 KB B panel (8 n-tiles) is DMA'd into LDS by the
// Tensor Data Mover (double-buffered), waves consume it via ds_load_b128.
// =====================================================================
__global__ __launch_bounds__(256) void gemm_bf16_wmma(const __bf16* __restrict__ Ap,
                                                      const __bf16* __restrict__ Bp,
                                                      float* __restrict__ C,
                                                      int M, int N, int K) {
  const int Nt = N >> 4, Kt = K >> 5;
  const int nbBlocks = N >> 7;                 // 128-column blocks
  int bm = blockIdx.x / nbBlocks;
  int bn = blockIdx.x % nbBlocks;
  int wid  = threadIdx.x >> 5;
  int lane = threadIdx.x & 31;
  int wm = wid >> 2;                           // 0..1  (row 32-patch in block)
  int wn = wid & 3;                            // 0..3  (col 32-patch in block)

  __shared__ __align__(32) __bf16 Bbuf[2][8 * 512];   // double-buffered 8 KB panels

  const v4i z4 = {0, 0, 0, 0};
  const v8i z8 = {0, 0, 0, 0, 0, 0, 0, 0};

  // ---- constant D# group1: data_size=8B, 1D tile of 1024 units (8 KB),
  //      tile_dim1=1, huge tensor_dim0 (no OOB clipping), stride0=tile.
  v8i g1;
  g1[0] = (3 << 16);                // workgroup_mask=0, data_size=3 (8B)
  g1[1] = 0;                        // atomic_barrier_addr=0, tensor_dim0 lo16=0
  g1[2] = 16 | (1 << 16);           // tensor_dim0 hi16 (2^20 units), tensor_dim1 lo16=1
  g1[3] = (1024 << 16);             // tensor_dim1 hi16=0, tile_dim0=1024
  g1[4] = 1;                        // tile_dim1=1, tile_dim2=0
  g1[5] = 1024;                     // tensor_dim0_stride lo32
  g1[6] = 0;
  g1[7] = 0;

  auto tdm_issue = [&](int kt, int sel) {
    unsigned long long ga =
        (unsigned long long)(const void*)Bp + ((size_t)(kt * Nt + bn * 8) * 512) * 2;
    unsigned ldsa = (unsigned)(size_t)(&Bbuf[sel][0]);
    v4u g0;
    g0[0] = 1u;                                        // count=1, user descriptor
    g0[1] = ldsa;                                      // lds_addr
    g0[2] = (unsigned)ga;                              // global_addr[31:0]
    g0[3] = (unsigned)((ga >> 32) & 0x1FFFFFFu) | 0x80000000u;  // addr[56:32] | type=2
    TDM_LOAD_TO_LDS(g0, g1, z4, z8);
  };

  // prologue: DMA first panel
  if (wid == 0) {
    tdm_issue(0, 0);
    __builtin_amdgcn_s_wait_tensorcnt(0);
  }
  __syncthreads();

  v8f c00 = {}, c01 = {}, c10 = {}, c11 = {};
  const __bf16* a0p = Ap + (size_t)(2 * (bm * 2 + wm) + 0) * Kt * 512 + lane * 16;
  const __bf16* a1p = Ap + (size_t)(2 * (bm * 2 + wm) + 1) * Kt * 512 + lane * 16;

  for (int kt = 0; kt < Kt; ++kt) {
    int sel = kt & 1;
    if (wid == 0 && (kt + 1) < Kt) tdm_issue(kt + 1, sel ^ 1);   // prefetch next panel

    v16bf a0 = *(const v16bf*)(a0p + (size_t)kt * 512);
    v16bf a1 = *(const v16bf*)(a1p + (size_t)kt * 512);
    __builtin_prefetch(a0p + (size_t)(kt + 1) * 512, 0, 0);

    const __bf16* bl = &Bbuf[sel][(2 * wn) * 512 + lane * 16];
    v16bf b0 = *(const v16bf*)(bl);
    v16bf b1 = *(const v16bf*)(bl + 512);

    c00 = WMMA_BF16(a0, b0, c00);
    c01 = WMMA_BF16(a0, b1, c01);
    c10 = WMMA_BF16(a1, b0, c10);
    c11 = WMMA_BF16(a1, b1, c11);

    if (wid == 0 && (kt + 1) < Kt) __builtin_amdgcn_s_wait_tensorcnt(0);
    __syncthreads();
  }

  int half = lane >> 4, n = lane & 15;
#pragma unroll
  for (int r = 0; r < 8; ++r) {
    int m0 = bm * 64 + wm * 32 + 8 * half + r;
    int m1 = m0 + 16;
    size_t o0 = (size_t)m0 * N + bn * 128 + wn * 32 + n;
    size_t o1 = (size_t)m1 * N + bn * 128 + wn * 32 + n;
    C[o0]      = c00[r];
    C[o0 + 16] = c01[r];
    C[o1]      = c10[r];
    C[o1 + 16] = c11[r];
  }
}

// =====================================================================
// sliding-window flash attention: one wave per (b, h, 16-query tile)
// =====================================================================
__global__ __launch_bounds__(32) void attn_kernel(const __bf16* __restrict__ Qp,
                                                  const __bf16* __restrict__ Kp,
                                                  const __bf16* __restrict__ Vp,
                                                  __bf16* __restrict__ Op) {
  __shared__ __bf16 Pl[16 * 32];

  int lane = threadIdx.x;
  int qt = blockIdx.x % (SS / 16);
  int h  = (blockIdx.x / (SS / 16)) % NH;
  int b  = blockIdx.x / ((SS / 16) * NH);
  int kvh = h / (NH / NKV);
  int i0 = qt * 16;
  int half = lane >> 4, nn = lane & 15;

  size_t qbase = ((size_t)(b * NH + h) * (SS / 16) + qt) * 2 * 512;
  v16bf qa0 = *(const v16bf*)(Qp + qbase +       (size_t)lane * 16);
  v16bf qa1 = *(const v16bf*)(Qp + qbase + 512 + (size_t)lane * 16);

  v8f o0 = {}, o1 = {}, o2 = {}, o3 = {};
  float mi[8], li[8];
#pragma unroll
  for (int r = 0; r < 8; ++r) { mi[r] = -1e30f; li[r] = 0.0f; }

  int jlo = i0 - (WIN - 1); if (jlo < 0) jlo = 0;
  int kt_lo = jlo >> 5;
  int kt_hi = (i0 + 15) >> 5;

  size_t kvbase = (size_t)(b * NKV + kvh);

  for (int kt = kt_lo; kt <= kt_hi; ++kt) {
    int kbase = kt << 5;
    size_t kb = (kvbase * (SS / 16) + 2 * kt) * 2 * 512 + (size_t)lane * 16;
    v16bf k00 = *(const v16bf*)(Kp + kb);
    v16bf k01 = *(const v16bf*)(Kp + kb + 512);
    v16bf k10 = *(const v16bf*)(Kp + kb + 1024);
    v16bf k11 = *(const v16bf*)(Kp + kb + 1536);
    v8f s0 = {}, s1 = {};
    s0 = WMMA_BF16(qa0, k00, s0);
    s0 = WMMA_BF16(qa1, k01, s0);
    s1 = WMMA_BF16(qa0, k10, s1);
    s1 = WMMA_BF16(qa1, k11, s1);

    float p0[8], p1[8], al[8];
#pragma unroll
    for (int r = 0; r < 8; ++r) {
      int i  = i0 + 8 * half + r;
      int j0 = kbase + nn;
      int j1 = j0 + 16;
      float v0 = ((j0 <= i) && (j0 > i - WIN)) ? s0[r] : -1e30f;
      float v1 = ((j1 <= i) && (j1 > i - WIN)) ? s1[r] : -1e30f;
      float mx = fmaxf(v0, v1);
#pragma unroll
      for (int off = 1; off < 16; off <<= 1) mx = fmaxf(mx, __shfl_xor(mx, off, 32));
      float mnew = fmaxf(mi[r], mx);
      float a  = __expf(mi[r] - mnew);
      float e0 = __expf(v0 - mnew);
      float e1 = __expf(v1 - mnew);
      float rs = e0 + e1;
#pragma unroll
      for (int off = 1; off < 16; off <<= 1) rs += __shfl_xor(rs, off, 32);
      li[r] = li[r] * a + rs;
      mi[r] = mnew;
      al[r] = a;
      p0[r] = e0;
      p1[r] = e1;
    }
#pragma unroll
    for (int r = 0; r < 8; ++r) {
      o0[r] *= al[r]; o1[r] *= al[r]; o2[r] *= al[r]; o3[r] *= al[r];
    }

    // P: C-layout -> A-layout via LDS
#pragma unroll
    for (int r = 0; r < 8; ++r) {
      int m = 8 * half + r;
      Pl[m * 32 + nn]      = f2bf(p0[r]);
      Pl[m * 32 + 16 + nn] = f2bf(p1[r]);
    }
    __syncthreads();
    union { v16bf v; __bf16 a[16]; } pa;
    int mrow = lane & 15;
#pragma unroll
    for (int e = 0; e < 16; ++e) {
      int k = e + (e >= 8 ? 8 : 0) + half * 8;
      pa.a[e] = Pl[mrow * 32 + k];
    }
    __syncthreads();

    size_t vb = (kvbase * (SS / 32) + kt) * 4 * 512 + (size_t)lane * 16;
    v16bf vf0 = *(const v16bf*)(Vp + vb);
    v16bf vf1 = *(const v16bf*)(Vp + vb + 512);
    v16bf vf2 = *(const v16bf*)(Vp + vb + 1024);
    v16bf vf3 = *(const v16bf*)(Vp + vb + 1536);
    o0 = WMMA_BF16(pa.v, vf0, o0);
    o1 = WMMA_BF16(pa.v, vf1, o1);
    o2 = WMMA_BF16(pa.v, vf2, o2);
    o3 = WMMA_BF16(pa.v, vf3, o3);
  }

  float inv[8];
#pragma unroll
  for (int r = 0; r < 8; ++r) inv[r] = 1.0f / li[r];
  int base_mt = (b * SS + i0) >> 4;
#pragma unroll
  for (int dt = 0; dt < 4; ++dt) {
    int col = h * HD + dt * 16 + nn;
    int ktile = col >> 5;
    int kk = col & 31;
    int ha, e;
    a_he(kk, ha, e);
    v8f oo = (dt == 0) ? o0 : (dt == 1) ? o1 : (dt == 2) ? o2 : o3;
#pragma unroll
    for (int r = 0; r < 8; ++r) {
      int mrow = 8 * half + r;
      size_t idx = (((size_t)base_mt * (DD / 32) + ktile) * 32 + ha * 16 + mrow) * 16 + e;
      Op[idx] = f2bf(oo[r] * inv[r]);
    }
  }
}

// =====================================================================
// host launcher
// =====================================================================
extern "C" void kernel_launch(void* const* d_in, const int* in_sizes, int n_in,
                              void* d_out, int out_size, void* d_ws, size_t ws_size,
                              hipStream_t stream) {
  const float* x   = (const float*)d_in[0];
  const float* wq  = (const float*)d_in[1];
  const float* wk  = (const float*)d_in[2];
  const float* wv  = (const float*)d_in[3];
  const float* wo  = (const float*)d_in[4];
  const float* qnw = (const float*)d_in[5];
  const float* knw = (const float*)d_in[6];
  float* out = (float*)d_out;

  char* p = (char*)d_ws;
  auto carve = [&](size_t bytes) -> char* {
    char* r = p;
    p += (bytes + 255) & ~(size_t)255;
    return r;
  };
  __bf16* Xp   = (__bf16*)carve((size_t)BS * DD * 2);
  __bf16* Wqp  = (__bf16*)carve((size_t)DD * DD * 2);
  __bf16* Wkp  = (__bf16*)carve((size_t)DD * (NKV * HD) * 2);
  __bf16* Wvp  = (__bf16*)carve((size_t)DD * (NKV * HD) * 2);
  __bf16* Wop  = (__bf16*)carve((size_t)DD * DD * 2);
  float*  qt   = (float*)carve((size_t)BS * DD * 4);
  float*  ktmp = (float*)carve((size_t)BS * (NKV * HD) * 4);
  float*  vtmp = (float*)carve((size_t)BS * (NKV * HD) * 4);
  __bf16* Qp   = (__bf16*)carve((size_t)BS * DD * 2);
  __bf16* Kp   = (__bf16*)carve((size_t)BB * NKV * SS * HD * 2);
  __bf16* Vp   = (__bf16*)carve((size_t)BB * NKV * SS * HD * 2);
  __bf16* Att  = (__bf16*)carve((size_t)BS * DD * 2);

  const int T = 256;

  // 1) pack inputs into WMMA fragment layouts (fp32 -> bf16)
  pack_a_kernel<<<(int)(((size_t)BS * DD + T - 1) / T), T, 0, stream>>>(x, Xp, BS, DD);
  pack_b_kernel<<<(int)(((size_t)DD * DD + T - 1) / T), T, 0, stream>>>(wq, Wqp, DD, DD);
  pack_b_kernel<<<(int)(((size_t)DD * NKV * HD + T - 1) / T), T, 0, stream>>>(wk, Wkp, DD, NKV * HD);
  pack_b_kernel<<<(int)(((size_t)DD * NKV * HD + T - 1) / T), T, 0, stream>>>(wv, Wvp, DD, NKV * HD);
  pack_b_kernel<<<(int)(((size_t)DD * DD + T - 1) / T), T, 0, stream>>>(wo, Wop, DD, DD);

  // 2) QKV projections (TDM-staged B panels, bf16 WMMA)
  gemm_bf16_wmma<<<(BS / 64) * (DD / 128), T, 0, stream>>>(Xp, Wqp, qt, BS, DD, DD);
  gemm_bf16_wmma<<<(BS / 64) * ((NKV * HD) / 128), T, 0, stream>>>(Xp, Wkp, ktmp, BS, NKV * HD, DD);
  gemm_bf16_wmma<<<(BS / 64) * ((NKV * HD) / 128), T, 0, stream>>>(Xp, Wvp, vtmp, BS, NKV * HD, DD);

  // 3) RMSNorm + RoPE + fragment packing (Q pre-scaled by HD^-0.5)
  norm_rope_pack_kernel<<<BB * SS * NH * 32 / T, T, 0, stream>>>(qt, qnw, Qp, NH, 1, 0.125f);
  norm_rope_pack_kernel<<<BB * SS * NKV * 32 / T, T, 0, stream>>>(ktmp, knw, Kp, NKV, 0, 1.0f);
  pack_v_kernel<<<(int)(((size_t)BB * NKV * SS * HD + T - 1) / T), T, 0, stream>>>(vtmp, Vp);

  // 4) sliding-window flash attention (WMMA QK^T and PV)
  attn_kernel<<<BB * NH * (SS / 16), 32, 0, stream>>>(Qp, Kp, Vp, Att);

  // 5) output projection
  gemm_bf16_wmma<<<(BS / 64) * (DD / 128), T, 0, stream>>>(Att, Wop, out, BS, DD, DD);
}
